// DRAGON_20761871909655
// MI455X (gfx1250) — matrix operand order repaired and checked by reading, not compiled, not run on hardware
//
#include <hip/hip_runtime.h>
#include <math.h>

typedef __attribute__((ext_vector_type(16))) _Float16 v16h;
typedef __attribute__((ext_vector_type(8)))  _Float16 v8h;
typedef __attribute__((ext_vector_type(8)))  float    v8f;
typedef __attribute__((ext_vector_type(4)))  float    v4f;

// ---------------- problem constants ----------------
constexpr int kB = 8, kS = 256, kSD = 768, kL = 12, kNH = 12, kDH = 64, kFF = 3072;
constexpr int kKL = 5, kD = 200, kHC = 4, kDPH = 50, kNNT = 4, kNET = 38, kNN = 200;
constexpr int kN = kB * kNN;        // 1600 nodes
constexpr int kE = 50000;
constexpr int kEN = kE + kN;        // 51600 edges incl self loops
constexpr int kIE = 200;
constexpr float kLN_EPS  = 1e-12f;
constexpr float kRBN     = 0.99999500003749979f;   // 1/sqrt(1+1e-5) (BN eval)
constexpr float kSCALE_Q = 0.14142135623730951f;   // 1/sqrt(50)
constexpr float kSCALE_A = 0.125f;                 // 1/sqrt(64)

// ---------------- generic WMMA GEMM ----------------
constexpr int GBM = 128, GBN = 64, GBK = 32, ASTR = 40, BSTR = 40;

struct GemmP {
  const float* A; const int* a_idx; long long a_outer, a_inner; int lda;
  const void*  B; int b_f16; long long b_outer, b_inner; int ldb_n, ldb_k;
  float* C; long long c_outer, c_inner; int ldc;
  const float* bias; long long bias_outer, bias_inner;
  const float* nscale; const float* nshift; float ns_mul;
  const float* resid; long long r_outer, r_inner; int ldr;
  int M, N, K, zdiv, act, accumulate;   // act: 0 none, 1 gelu(exact), 2 relu
  float alpha;
};

// C = epilogue(alpha * A(idx) @ B + ...) using v_wmma_f32_16x16x32_f16.
// A: f32 row-major [M][lda] (optional row gather). B: logically [N][K] with
// strides (ldb_n, ldb_k); f16 (pre-transposed weights) or f32 (activations).
// Interior blocks with f16 unit-stride B take a double-buffered unguarded
// path (1 barrier / K-step, global loads overlap WMMA, hoisted gathers,
// global_prefetch for tile+2); everything else takes the guarded path.
__global__ __launch_bounds__(256) void glm_gemm_wmma(GemmP p) {
  __shared__ _Float16 As[2][GBM * ASTR];
  __shared__ _Float16 Bs[2][GBN * BSTR];
  int z  = blockIdx.z;
  int zo = z / p.zdiv, zi = z - zo * p.zdiv;
  const float* A = p.A + zo * p.a_outer + zi * p.a_inner;
  const _Float16* Bh = nullptr; const float* Bf = nullptr;
  if (p.b_f16) Bh = (const _Float16*)p.B + zo * p.b_outer + zi * p.b_inner;
  else         Bf = (const float*)p.B    + zo * p.b_outer + zi * p.b_inner;
  float* C = p.C + zo * p.c_outer + zi * p.c_inner;

  int m0 = blockIdx.y * GBM, n0 = blockIdx.x * GBN;
  int t = threadIdx.x, lane = t & 31, wave = t >> 5;
  int wm = (wave & 3) * 32, wn = (wave >> 2) * 32;
  v8f acc[2][2] = {};

  // fragment load + 2x2 WMMA from one staged tile (CDNA5 wave32 layouts)
  auto compute = [&](const _Float16* as, const _Float16* bs) {
    v16h af[2], bfr[2];
#pragma unroll
    for (int ti = 0; ti < 2; ++ti) {
      int row = wm + ti * 16 + (lane & 15);
      int hb = (lane >> 4) << 3;
      v8h lo = *(const v8h*)&as[row * ASTR + hb];
      v8h hi = *(const v8h*)&as[row * ASTR + 16 + hb];
#pragma unroll
      for (int h = 0; h < 8; ++h) { af[ti][h] = lo[h]; af[ti][h + 8] = hi[h]; }
    }
#pragma unroll
    for (int tj = 0; tj < 2; ++tj) {
      int nr = wn + tj * 16 + (lane & 15);
      int kb = (lane >> 4) << 3;
      v8h lo = *(const v8h*)&bs[nr * BSTR + kb];
      v8h hi = *(const v8h*)&bs[nr * BSTR + kb + 16];
#pragma unroll
      for (int h = 0; h < 8; ++h) { bfr[tj][h] = lo[h]; bfr[tj][h + 8] = hi[h]; }
    }
#pragma unroll
    for (int ti = 0; ti < 2; ++ti)
#pragma unroll
      for (int tj = 0; tj < 2; ++tj)
        acc[ti][tj] = __builtin_amdgcn_wmma_f32_16x16x32_f16(
            false, af[ti], false, bfr[tj], (short)0, acc[ti][tj], false, false);
  };

  bool fast = (m0 + GBM <= p.M) && (n0 + GBN <= p.N) && p.b_f16 && (p.ldb_k == 1);
  if (fast) {
    // hoist row gathers / base pointers out of the K loop
    const float* aRow[4];
#pragma unroll
    for (int i = 0; i < 4; ++i) {
      int idx = i * 256 + t;
      int r = idx >> 3, c4 = (idx & 7) << 2;
      long long ar = p.a_idx ? (long long)p.a_idx[m0 + r] : (long long)(m0 + r);
      aRow[i] = A + ar * p.lda + c4;
    }
    const _Float16* bRow = Bh + (long long)(n0 + (t >> 2)) * p.ldb_n + ((t & 3) << 3);
    int nT = (p.K + GBK - 1) / GBK;

    auto stage = [&](int tile, _Float16* as, _Float16* bs) {
      int k0 = tile * GBK;
      if (k0 + GBK <= p.K) {                        // full tile: unguarded
#pragma unroll
        for (int i = 0; i < 4; ++i) {
          int idx = i * 256 + t;
          int r = idx >> 3, c4 = (idx & 7) << 2;
          v4f vals = *(const v4f*)(aRow[i] + k0);
#pragma unroll
          for (int j = 0; j < 4; ++j) as[r * ASTR + c4 + j] = (_Float16)vals[j];
        }
        v8h vv = *(const v8h*)(bRow + k0);
        *(v8h*)&bs[(t >> 2) * BSTR + ((t & 3) << 3)] = vv;
      } else {                                      // K tail: guard k only
#pragma unroll
        for (int i = 0; i < 4; ++i) {
          int idx = i * 256 + t;
          int r = idx >> 3, c4 = (idx & 7) << 2;
#pragma unroll
          for (int j = 0; j < 4; ++j) {
            int kg = k0 + c4 + j;
            as[r * ASTR + c4 + j] = (kg < p.K) ? (_Float16)aRow[i][k0 + j] : (_Float16)0.0f;
          }
        }
        int kk = (t & 3) << 3;
#pragma unroll
        for (int j = 0; j < 8; ++j) {
          int kg = k0 + kk + j;
          bs[(t >> 2) * BSTR + kk + j] = (kg < p.K) ? bRow[k0 + j] : (_Float16)0.0f;
        }
      }
    };

    stage(0, As[0], Bs[0]);
    __syncthreads();
    int buf = 0;
    for (int it = 0; it < nT; ++it) {
      if (it + 1 < nT) {
        stage(it + 1, As[buf ^ 1], Bs[buf ^ 1]);
        __builtin_prefetch(aRow[0] + (it + 2) * GBK, 0, 1);  // global_prefetch_b8
        __builtin_prefetch(bRow + (it + 2) * GBK, 0, 1);
      }
      compute(As[buf], Bs[buf]);
      __syncthreads();
      buf ^= 1;
    }
  } else {
    for (int k0 = 0; k0 < p.K; k0 += GBK) {
      // ---- guarded A staging (f32 -> f16), 128x32 ----
#pragma unroll
      for (int i = 0; i < 4; ++i) {
        int idx = i * 256 + t;
        int r = idx >> 3, c4 = (idx & 7) << 2;
        int gr = m0 + r;
        v4f vals = {0.f, 0.f, 0.f, 0.f};
        if (gr < p.M) {
          long long ar = p.a_idx ? (long long)p.a_idx[gr] : (long long)gr;
          const float* ap = A + ar * p.lda + k0 + c4;
          if (k0 + c4 + 4 <= p.K) {
            vals = *(const v4f*)ap;
          } else {
#pragma unroll
            for (int j = 0; j < 4; ++j) if (k0 + c4 + j < p.K) vals[j] = ap[j];
          }
        }
#pragma unroll
        for (int j = 0; j < 4; ++j) As[0][r * ASTR + c4 + j] = (_Float16)vals[j];
      }
      // ---- guarded B staging, N-major [64][32] ----
      {
        int n = t >> 2, kk = (t & 3) << 3;
        int gn = n0 + n, gk = k0 + kk;
        if (p.b_f16 && p.ldb_k == 1 && gn < p.N && gk + 8 <= p.K) {
          v8h vv = *(const v8h*)(Bh + (long long)gn * p.ldb_n + gk);
          *(v8h*)&Bs[0][n * BSTR + kk] = vv;
        } else {
#pragma unroll
          for (int j = 0; j < 8; ++j) {
            float v = 0.0f; int k = gk + j;
            if (gn < p.N && k < p.K) {
              long long boff = (long long)gn * p.ldb_n + (long long)k * p.ldb_k;
              v = p.b_f16 ? (float)Bh[boff] : Bf[boff];
            }
            Bs[0][n * BSTR + kk + j] = (_Float16)v;
          }
        }
      }
      __syncthreads();
      compute(As[0], Bs[0]);
      __syncthreads();
    }
  }
  // ---- fused epilogue ----
  const float* bias  = p.bias  ? p.bias  + zo * p.bias_outer + zi * p.bias_inner : nullptr;
  const float* resid = p.resid ? p.resid + zo * p.r_outer    + zi * p.r_inner    : nullptr;
#pragma unroll
  for (int ti = 0; ti < 2; ++ti) {
#pragma unroll
    for (int tj = 0; tj < 2; ++tj) {
      int gn  = n0 + wn + tj * 16 + (lane & 15);
      int gmb = m0 + wm + ti * 16 + ((lane >> 4) << 3);
      if (gn < p.N) {
#pragma unroll
        for (int r = 0; r < 8; ++r) {
          int gm = gmb + r;
          if (gm < p.M) {
            float v = p.alpha * acc[ti][tj][r];
            float* cp = C + (long long)gm * p.ldc + gn;
            if (p.accumulate) v += *cp;
            if (bias)      v += bias[gn];
            if (p.nscale)  v  = v * p.nscale[gn] * p.ns_mul;
            if (p.nshift)  v += p.nshift[gn];
            if (resid)     v += resid[(long long)gm * p.ldr + gn];
            if (p.act == 1)      v = 0.5f * v * (1.0f + erff(v * 0.70710678118654752f));
            else if (p.act == 2) v = fmaxf(v, 0.0f);
            *cp = v;
          }
        }
      }
    }
  }
}

// ---------------- support kernels ----------------
__global__ void glm_transpose_f16(const float* in, _Float16* out, int Kd, int Nd) {
  long long boff = (long long)blockIdx.z * Kd * Nd;
  in += boff; out += boff;
  int n  = blockIdx.x * 32 + (threadIdx.x & 31);
  int k0 = blockIdx.y * 32 + (threadIdx.x >> 5);
  for (int i = 0; i < 4; ++i) {
    int k = k0 + i * 8;
    if (k < Kd && n < Nd) out[(long long)n * Kd + k] = (_Float16)in[(long long)k * Nd + n];
  }
}

__global__ __launch_bounds__(256) void glm_layernorm(const float* x, const float* w,
                                                     const float* b, float* y, int cols) {
  __shared__ float red[256];
  long long row = blockIdx.x;
  const float* xr = x + row * cols;
  float* yr = y + row * cols;
  int t = threadIdx.x;
  float s = 0.f;
  for (int c = t; c < cols; c += 256) s += xr[c];
  red[t] = s; __syncthreads();
  for (int o = 128; o > 0; o >>= 1) { if (t < o) red[t] += red[t + o]; __syncthreads(); }
  float mean = red[0] / cols;
  __syncthreads();
  float vs = 0.f;
  for (int c = t; c < cols; c += 256) { float d = xr[c] - mean; vs += d * d; }
  red[t] = vs; __syncthreads();
  for (int o = 128; o > 0; o >>= 1) { if (t < o) red[t] += red[t + o]; __syncthreads(); }
  float rstd = rsqrtf(red[0] / cols + kLN_EPS);
  for (int c = t; c < cols; c += 256) yr[c] = (xr[c] - mean) * rstd * w[c] + b[c];
}

__global__ __launch_bounds__(256) void glm_softmax_rows(float* x, int cols) {
  __shared__ float red[256];
  long long row = blockIdx.x;
  float* xr = x + row * cols;
  int t = threadIdx.x;
  float m = -3.0e38f;
  for (int c = t; c < cols; c += 256) m = fmaxf(m, xr[c]);
  red[t] = m; __syncthreads();
  for (int o = 128; o > 0; o >>= 1) { if (t < o) red[t] = fmaxf(red[t], red[t + o]); __syncthreads(); }
  m = red[0]; __syncthreads();
  float s = 0.f;
  for (int c = t; c < cols; c += 256) { float e = expf(xr[c] - m); xr[c] = e; s += e; }
  red[t] = s; __syncthreads();
  for (int o = 128; o > 0; o >>= 1) { if (t < o) red[t] += red[t + o]; __syncthreads(); }
  float inv = 1.0f / red[0];
  __syncthreads();
  for (int c = t; c < cols; c += 256) xr[c] *= inv;
}

__global__ void glm_fill(float* p, float v, long long n) {
  long long i = (long long)blockIdx.x * 256 + threadIdx.x;
  if (i < n) p[i] = v;
}

__global__ void glm_build_edges(const int* eidx, int* srcF, int* tgtF) {
  int e = blockIdx.x * 256 + threadIdx.x;
  if (e >= kEN) return;
  if (e < kE) { srcF[e] = eidx[e]; tgtF[e] = eidx[kE + e]; }
  else        { srcF[e] = e - kE;  tgtF[e] = e - kE; }
}

__global__ void glm_deg(const int* srcF, float* deg) {
  int e = blockIdx.x * 256 + threadIdx.x;
  if (e < kEN) atomicAdd(&deg[srcF[e]], 1.0f);
}

// feat@ee_w1 is a 3-row gather (one-hot matmul) + bias + BN(eval) + relu
__global__ void glm_edge_pre(const float* w1, const float* b1, const float* bnw,
                             const float* bnb, const int* etype, const int* ntype,
                             const int* srcF, const int* tgtF, float* h) {
  long long i = (long long)blockIdx.x * 256 + threadIdx.x;
  if (i >= (long long)kEN * kD) return;
  int e = (int)(i / kD), d = (int)(i % kD);
  int r0 = (e < kE) ? etype[e] : kNET;
  int st = ntype[srcF[e]], tt = ntype[tgtF[e]];
  float v = w1[r0 * kD + d] + w1[(kNET + 1 + st) * kD + d]
          + w1[(kNET + 1 + kNNT + tt) * kD + d] + b1[d];
  v = v * (kRBN * bnw[d]) + bnb[d];
  h[i] = fmaxf(v, 0.0f);
}

__global__ void glm_concat2(const float* Xc, const float* nfe, float* x2) {
  long long i = (long long)blockIdx.x * 256 + threadIdx.x;
  if (i >= (long long)kN * 2 * kD) return;
  int r = (int)(i / (2 * kD)), c = (int)(i % (2 * kD));
  x2[i] = (c < kD) ? Xc[(long long)r * kD + c] : nfe[(long long)r * kD + (c - kD)];
}

__global__ void glm_gat_scores(const float* qry, const float* key, float* esc) {
  int i = blockIdx.x * 256 + threadIdx.x;
  if (i >= kEN * kHC) return;
  int e = i >> 2, hc = i & 3;
  const float* q = qry + (long long)e * kD + hc * kDPH;
  const float* k = key + (long long)e * kD + hc * kDPH;
  float s = 0.f;
  for (int d = 0; d < kDPH; ++d) s += q[d] * k[d];
  esc[i] = s * kSCALE_Q;
}

__device__ inline void glm_atomic_max_f(float* addr, float val) {
  unsigned* ua = (unsigned*)addr;
  unsigned old = *ua;
  for (;;) {
    float f = __uint_as_float(old);
    if (f >= val) return;
    unsigned assumed = old;
    old = atomicCAS(ua, assumed, __float_as_uint(val));
    if (old == assumed) return;
  }
}

__global__ void glm_seg_max(const float* esc, const int* srcF, float* smax) {
  int i = blockIdx.x * 256 + threadIdx.x;
  if (i >= kEN * kHC) return;
  glm_atomic_max_f(&smax[srcF[i >> 2] * kHC + (i & 3)], esc[i]);
}

__global__ void glm_seg_exp(float* esc, const int* srcF, const float* smax, float* den) {
  int i = blockIdx.x * 256 + threadIdx.x;
  if (i >= kEN * kHC) return;
  int s = srcF[i >> 2] * kHC + (i & 3);
  float e = expf(esc[i] - smax[s]);
  esc[i] = e;
  atomicAdd(&den[s], e);
}

__global__ void glm_seg_alpha(const float* esc, const int* srcF, const float* den,
                              const float* deg, float* ealpha) {
  int i = blockIdx.x * 256 + threadIdx.x;
  if (i >= kEN * kHC) return;
  int sn = srcF[i >> 2];
  ealpha[i] = esc[i] / (den[sn * kHC + (i & 3)] + 1e-16f) * deg[sn];
}

__global__ void glm_aggregate(const float* msg, const float* ealpha, const int* tgtF, float* aggr) {
  long long i = (long long)blockIdx.x * 256 + threadIdx.x;
  if (i >= (long long)kEN * kD) return;
  int e = (int)(i / kD), d = (int)(i % kD);
  float v = msg[i] * ealpha[e * kHC + d / kDPH];
  atomicAdd(&aggr[(long long)tgtF[e] * kD + d], v);
}

__global__ void glm_ie_gather(const float* hs, const float* Xc, float* cf) {
  int i = blockIdx.x * 256 + threadIdx.x;
  if (i >= kB * (kSD + kD)) return;
  int b = i / (kSD + kD), j = i % (kSD + kD);
  cf[i] = (j < kSD) ? hs[(long long)b * kS * kSD + j]
                    : Xc[(long long)b * kNN * kD + (j - kSD)];
}

__global__ void glm_ie_scatter(const float* cf2, float* hs, float* Xc) {
  int i = blockIdx.x * 256 + threadIdx.x;
  if (i >= kB * (kSD + kD)) return;
  int b = i / (kSD + kD), j = i % (kSD + kD);
  if (j < kSD) hs[(long long)b * kS * kSD + j] = cf2[i];
  else         Xc[(long long)b * kNN * kD + (j - kSD)] = cf2[i];
}

// ---------------- host helpers ----------------
static GemmP mk(const float* A, int lda, const void* B, int b_f16, int ldb_n, int ldb_k,
                float* C, int ldc, int M, int N, int K) {
  GemmP p{};
  p.A = A; p.lda = lda; p.B = B; p.b_f16 = b_f16; p.ldb_n = ldb_n; p.ldb_k = ldb_k;
  p.C = C; p.ldc = ldc; p.M = M; p.N = N; p.K = K;
  p.zdiv = 1; p.alpha = 1.0f; p.ns_mul = 1.0f;
  return p;
}

static void glm_gemm(hipStream_t s, const GemmP& p, int batches) {
  dim3 grid((p.N + GBN - 1) / GBN, (p.M + GBM - 1) / GBM, batches);
  glm_gemm_wmma<<<grid, 256, 0, s>>>(p);
}

static inline int cdiv(long long a, int b) { return (int)((a + b - 1) / b); }

// ---------------- orchestration ----------------
extern "C" void kernel_launch(void* const* d_in, const int* in_sizes, int n_in,
                              void* d_out, int out_size, void* d_ws, size_t ws_size,
                              hipStream_t stream) {
  (void)in_sizes; (void)n_in; (void)out_size; (void)ws_size;
  const float* hidden = (const float*)d_in[0];
  const float* amask  = (const float*)d_in[1];
  const float* X0     = (const float*)d_in[2];
  const float* nfe    = (const float*)d_in[3];
  const float* qkv_w  = (const float*)d_in[4];
  const float* qkv_b  = (const float*)d_in[5];
  const float* ao_w   = (const float*)d_in[6];
  const float* ao_b   = (const float*)d_in[7];
  const float* ln1w   = (const float*)d_in[8];
  const float* ln1b   = (const float*)d_in[9];
  const float* f1w    = (const float*)d_in[10];
  const float* f1b    = (const float*)d_in[11];
  const float* f2w    = (const float*)d_in[12];
  const float* f2b    = (const float*)d_in[13];
  const float* ln2w   = (const float*)d_in[14];
  const float* ln2b   = (const float*)d_in[15];
  const float* eew1   = (const float*)d_in[16];
  const float* eeb1   = (const float*)d_in[17];
  const float* eebnw  = (const float*)d_in[18];
  const float* eebnb  = (const float*)d_in[19];
  const float* eew2   = (const float*)d_in[20];
  const float* eeb2   = (const float*)d_in[21];
  const float* gkw    = (const float*)d_in[22];
  const float* gkb    = (const float*)d_in[23];
  const float* gmw    = (const float*)d_in[24];
  const float* gmb    = (const float*)d_in[25];
  const float* gqw    = (const float*)d_in[26];
  const float* gqb    = (const float*)d_in[27];
  const float* m1w    = (const float*)d_in[28];
  const float* m1b    = (const float*)d_in[29];
  const float* mbnw   = (const float*)d_in[30];
  const float* mbnb   = (const float*)d_in[31];
  const float* m2w    = (const float*)d_in[32];
  const float* m2b    = (const float*)d_in[33];
  const float* iew1   = (const float*)d_in[34];
  const float* ieb1   = (const float*)d_in[35];
  const float* iew2   = (const float*)d_in[36];
  const float* ieb2   = (const float*)d_in[37];
  const int* eidx     = (const int*)d_in[38];
  const int* etype    = (const int*)d_in[39];
  const int* ntype    = (const int*)d_in[40];

  // ---- workspace bump allocator ----
  char* base = (char*)d_ws;
  size_t off = 0;
  auto alloc = [&](size_t nbytes) -> void* {
    off = (off + 255) & ~(size_t)255;
    void* p = base + off;
    off += nbytes;
    return p;
  };
  // f16 transposed weights ([N][K] layout)
  _Float16* qkvT  = (_Float16*)alloc((size_t)kL * 3 * kSD * kSD * 2);
  _Float16* aoT   = (_Float16*)alloc((size_t)kL * kSD * kSD * 2);
  _Float16* f1T   = (_Float16*)alloc((size_t)kL * kSD * kFF * 2);
  _Float16* f2T   = (_Float16*)alloc((size_t)kL * kFF * kSD * 2);
  _Float16* eew2T = (_Float16*)alloc((size_t)kD * kD * 2);
  _Float16* gkT   = (_Float16*)alloc((size_t)kKL * 3 * kD * kD * 2);  // [200][600]
  _Float16* gmT   = (_Float16*)alloc((size_t)kKL * 3 * kD * kD * 2);
  _Float16* gqT   = (_Float16*)alloc((size_t)kKL * 2 * kD * kD * 2);  // [200][400]
  _Float16* m1T   = (_Float16*)alloc((size_t)kKL * kD * kD * 2);
  _Float16* m2T   = (_Float16*)alloc((size_t)kKL * kD * kD * 2);
  _Float16* iew1T = (_Float16*)alloc((size_t)(kSD + kD) * kIE * 2);
  _Float16* iew2T = (_Float16*)alloc((size_t)kIE * (kSD + kD) * 2);
  // f32 activations
  float* qkvout = (float*)alloc((size_t)kB * kS * 3 * kSD * 4);
  float* scores = (float*)alloc((size_t)kB * kNH * kS * kS * 4);  // aliased as FFN inter
  float* ctx    = (float*)alloc((size_t)kB * kS * kSD * 4);
  float* tmp    = (float*)alloc((size_t)kB * kS * kSD * 4);
  float* attnb  = (float*)alloc((size_t)kB * kS * kSD * 4);
  float* hbuf   = (float*)alloc((size_t)kEN * kD * 4);            // aliased as keyb
  float* eemb   = (float*)alloc((size_t)kEN * kD * 4);
  float* msgb   = (float*)alloc((size_t)kEN * kD * 4);
  float* qryb   = (float*)alloc((size_t)kEN * kD * 4);
  float* esc    = (float*)alloc((size_t)kEN * kHC * 4);
  float* ealpha = (float*)alloc((size_t)kEN * kHC * 4);
  float* smax   = (float*)alloc((size_t)kN * kHC * 4);
  float* den    = (float*)alloc((size_t)kN * kHC * 4);
  float* deg    = (float*)alloc((size_t)kN * 4);
  float* x2     = (float*)alloc((size_t)kN * 2 * kD * 4);
  float* aggr   = (float*)alloc((size_t)kN * kD * 4);
  float* hm     = (float*)alloc((size_t)kN * kD * 4);
  float* cf     = (float*)alloc((size_t)kB * (kSD + kD) * 4);
  float* cfh    = (float*)alloc((size_t)kB * kIE * 4);
  float* cf2    = (float*)alloc((size_t)kB * (kSD + kD) * 4);
  int* srcF     = (int*)alloc((size_t)kEN * 4);
  int* tgtF     = (int*)alloc((size_t)kEN * 4);
  float* inter  = scores;   // safe: disjoint lifetimes within a layer
  float* keyb   = hbuf;     // safe: hbuf dead after edge_emb GEMM

  // hs / Xc live directly in d_out (hs first, Xc after; matches tuple order)
  float* hs = (float*)d_out;
  float* Xc = hs + (size_t)kB * kS * kSD;
  hipMemcpyAsync(hs, hidden, (size_t)kB * kS * kSD * 4, hipMemcpyDeviceToDevice, stream);
  hipMemcpyAsync(Xc, X0, (size_t)kN * kD * 4, hipMemcpyDeviceToDevice, stream);

  // ---- one-shot weight transpose + f16 downconvert ----
  auto tl = [&](const float* in, _Float16* out, int Kd, int Nd, int batch) {
    dim3 g((Nd + 31) / 32, (Kd + 31) / 32, batch);
    glm_transpose_f16<<<g, 256, 0, stream>>>(in, out, Kd, Nd);
  };
  tl(qkv_w, qkvT, kSD, kSD, kL * 3);
  tl(ao_w,  aoT,  kSD, kSD, kL);
  tl(f1w,   f1T,  kSD, kFF, kL);
  tl(f2w,   f2T,  kFF, kSD, kL);
  tl(eew2,  eew2T, kD, kD, 1);
  tl(gkw,   gkT, 3 * kD, kD, kKL);
  tl(gmw,   gmT, 3 * kD, kD, kKL);
  tl(gqw,   gqT, 2 * kD, kD, kKL);
  tl(m1w,   m1T, kD, kD, kKL);
  tl(m2w,   m2T, kD, kD, kKL);
  tl(iew1,  iew1T, kSD + kD, kIE, 1);
  tl(iew2,  iew2T, kIE, kSD + kD, 1);

  // ---- graph preprocessing (shared across all GNN layers) ----
  glm_build_edges<<<cdiv(kEN, 256), 256, 0, stream>>>(eidx, srcF, tgtF);
  glm_fill<<<cdiv(kN, 256), 256, 0, stream>>>(deg, 0.f, kN);
  glm_deg<<<cdiv(kEN, 256), 256, 0, stream>>>(srcF, deg);
  glm_edge_pre<<<cdiv((long long)kEN * kD, 256), 256, 0, stream>>>(
      eew1, eeb1, eebnw, eebnb, etype, ntype, srcF, tgtF, hbuf);
  {
    GemmP p = mk(hbuf, kD, eew2T, 1, kD, 1, eemb, kD, kEN, kD, kD);
    p.bias = eeb2;
    glm_gemm(stream, p, 1);   // edge_emb = h @ ee_w2 + ee_b2
  }

  // ---- main loop: 12 transformer layers, GAT in the last 5 ----
  for (int l = 0; l < kL; ++l) {
    // QKV: [2048,768] @ f16 [2304][768] (q|k|v stacked along N)
    {
      GemmP p = mk(hs, kSD, qkvT + (size_t)l * 3 * kSD * kSD, 1, kSD, 1,
                   qkvout, 3 * kSD, kB * kS, 3 * kSD, kSD);
      p.bias = qkv_b + (size_t)l * 3 * kSD;
      glm_gemm(stream, p, 1);
    }
    // scores = scale * Q@K^T + mask   (96 batched 256x256x64 WMMA GEMMs)
    {
      GemmP p = mk(qkvout, 3 * kSD, qkvout + kSD, 0, 3 * kSD, 1, scores, kS, kS, kS, kDH);
      p.zdiv = kNH;
      p.a_outer = (long long)kS * 3 * kSD; p.a_inner = kDH;
      p.b_outer = (long long)kS * 3 * kSD; p.b_inner = kDH;
      p.c_outer = (long long)kNH * kS * kS; p.c_inner = (long long)kS * kS;
      p.alpha = kSCALE_A;
      p.bias = amask; p.bias_outer = kS; p.bias_inner = 0;
      glm_gemm(stream, p, kB * kNH);
    }
    glm_softmax_rows<<<kB * kNH * kS, 256, 0, stream>>>(scores, kS);
    // ctx = att @ V  (B fetched strided from qkvout)
    {
      GemmP p = mk(scores, kS, qkvout + 2 * kSD, 0, 1, 3 * kSD, ctx, kSD, kS, kDH, kS);
      p.zdiv = kNH;
      p.a_outer = (long long)kNH * kS * kS; p.a_inner = (long long)kS * kS;
      p.b_outer = (long long)kS * 3 * kSD;  p.b_inner = kDH;
      p.c_outer = (long long)kS * kSD;      p.c_inner = kDH;
      glm_gemm(stream, p, kB * kNH);
    }
    // attention out proj + residual, then LN1
    {
      GemmP p = mk(ctx, kSD, aoT + (size_t)l * kSD * kSD, 1, kSD, 1,
                   tmp, kSD, kB * kS, kSD, kSD);
      p.bias = ao_b + (size_t)l * kSD;
      p.resid = hs; p.ldr = kSD;
      glm_gemm(stream, p, 1);
    }
    glm_layernorm<<<kB * kS, 256, 0, stream>>>(tmp, ln1w + (size_t)l * kSD,
                                               ln1b + (size_t)l * kSD, attnb, kSD);
    // FFN1 (gelu fused) -> FFN2 (+residual) -> LN2 -> hs
    {
      GemmP p = mk(attnb, kSD, f1T + (size_t)l * kSD * kFF, 1, kSD, 1,
                   inter, kFF, kB * kS, kFF, kSD);
      p.bias = f1b + (size_t)l * kFF; p.act = 1;
      glm_gemm(stream, p, 1);
    }
    {
      GemmP p = mk(inter, kFF, f2T + (size_t)l * kFF * kSD, 1, kFF, 1,
                   tmp, kSD, kB * kS, kSD, kFF);
      p.bias = f2b + (size_t)l * kSD;
      p.resid = attnb; p.ldr = kSD;
      glm_gemm(stream, p, 1);
    }
    glm_layernorm<<<kB * kS, 256, 0, stream>>>(tmp, ln2w + (size_t)l * kSD,
                                               ln2b + (size_t)l * kSD, hs, kSD);

    if (l >= kL - kKL) {
      int g = l - (kL - kKL);
      glm_concat2<<<cdiv((long long)kN * 2 * kD, 256), 256, 0, stream>>>(Xc, nfe, x2);
      // key = [x2[tgt] | edge_emb] @ Wk + bk  (split-K accumulate, fused gather)
      {
        GemmP p = mk(x2, 2 * kD, gkT + (size_t)g * kD * 3 * kD, 1, 3 * kD, 1,
                     keyb, kD, kEN, kD, 2 * kD);
        p.a_idx = tgtF;
        glm_gemm(stream, p, 1);
        p = mk(eemb, kD, gkT + (size_t)g * kD * 3 * kD + 2 * kD, 1, 3 * kD, 1,
               keyb, kD, kEN, kD, kD);
        p.accumulate = 1; p.bias = gkb + (size_t)g * kD;
        glm_gemm(stream, p, 1);
      }
      // msg = [x2[src] | edge_emb] @ Wm + bm
      {
        GemmP p = mk(x2, 2 * kD, gmT + (size_t)g * kD * 3 * kD, 1, 3 * kD, 1,
                     msgb, kD, kEN, kD, 2 * kD);
        p.a_idx = srcF;
        glm_gemm(stream, p, 1);
        p = mk(eemb, kD, gmT + (size_t)g * kD * 3 * kD + 2 * kD, 1, 3 * kD, 1,
               msgb, kD, kEN, kD, kD);
        p.accumulate = 1; p.bias = gmb + (size_t)g * kD;
        glm_gemm(stream, p, 1);
      }
      // qry = x2[src] @ Wq + bq (scale folded into score kernel)
      {
        GemmP p = mk(x2, 2 * kD, gqT + (size_t)g * kD * 2 * kD, 1, 2 * kD, 1,
                     qryb, kD, kEN, kD, 2 * kD);
        p.a_idx = srcF; p.bias = gqb + (size_t)g * kD;
        glm_gemm(stream, p, 1);
      }
      // segment softmax over src + degree rescale, then scatter-aggregate to tgt
      glm_gat_scores<<<cdiv((long long)kEN * kHC, 256), 256, 0, stream>>>(qryb, keyb, esc);
      glm_fill<<<cdiv(kN * kHC, 256), 256, 0, stream>>>(smax, -3.0e38f, kN * kHC);
      glm_fill<<<cdiv(kN * kHC, 256), 256, 0, stream>>>(den, 0.f, kN * kHC);
      glm_seg_max<<<cdiv((long long)kEN * kHC, 256), 256, 0, stream>>>(esc, srcF, smax);
      glm_seg_exp<<<cdiv((long long)kEN * kHC, 256), 256, 0, stream>>>(esc, srcF, smax, den);
      glm_seg_alpha<<<cdiv((long long)kEN * kHC, 256), 256, 0, stream>>>(esc, srcF, den, deg, ealpha);
      glm_fill<<<cdiv((long long)kN * kD, 256), 256, 0, stream>>>(aggr, 0.f, (long long)kN * kD);
      glm_aggregate<<<cdiv((long long)kEN * kD, 256), 256, 0, stream>>>(msgb, ealpha, tgtF, aggr);
      // node MLP: BN(eval)+relu fused into GEMM1 epilogue; gelu fused into GEMM2
      {
        GemmP p = mk(aggr, kD, m1T + (size_t)g * kD * kD, 1, kD, 1, hm, kD, kN, kD, kD);
        p.bias = m1b + (size_t)g * kD;
        p.nscale = mbnw + (size_t)g * kD; p.nshift = mbnb + (size_t)g * kD;
        p.ns_mul = kRBN; p.act = 2;
        glm_gemm(stream, p, 1);
        p = mk(hm, kD, m2T + (size_t)g * kD * kD, 1, kD, 1, Xc, kD, kN, kD, kD);
        p.bias = m2b + (size_t)g * kD; p.act = 1;   // Xc = gelu(gat_layer(...))
        glm_gemm(stream, p, 1);
      }
      // info exchange between hs[:,0,:] and Xc[:,0,:]
      glm_ie_gather<<<cdiv(kB * (kSD + kD), 256), 256, 0, stream>>>(hs, Xc, cf);
      {
        GemmP p = mk(cf, kSD + kD, iew1T, 1, kSD + kD, 1, cfh, kIE, kB, kIE, kSD + kD);
        p.bias = ieb1; p.act = 1;
        glm_gemm(stream, p, 1);
        p = mk(cfh, kIE, iew2T, 1, kIE, 1, cf2, kSD + kD, kB, kSD + kD, kIE);
        p.bias = ieb2;
        glm_gemm(stream, p, 1);
      }
      glm_ie_scatter<<<cdiv(kB * (kSD + kD), 256), 256, 0, stream>>>(cf2, hs, Xc);
    }
  }
}
